// TransformerLayer_13520557047975
// MI455X (gfx1250) — compile-verified, matrix operand order checked by reading, not run on hardware
//
#include <hip/hip_runtime.h>
#include <hip/hip_bf16.h>

// ---------------------------------------------------------------------------
// Types / constants
// ---------------------------------------------------------------------------
typedef __bf16 bf16_t;
typedef __attribute__((ext_vector_type(8)))  bf16_t       v8bf;
typedef __attribute__((ext_vector_type(16))) bf16_t       v16bf;
typedef __attribute__((ext_vector_type(8)))  float        v8f;
typedef __attribute__((ext_vector_type(4)))  float        v4f;
typedef __attribute__((ext_vector_type(4)))  unsigned int u32x4;
typedef __attribute__((ext_vector_type(8)))  int          i32x8;
typedef __attribute__((ext_vector_type(4)))  int          i32x4;

#define DEV __device__ __forceinline__

constexpr int B_   = 2;
constexpr int S_   = 2048;
constexpr int D_   = 2048;
constexpr int HQ_  = 16;
constexpr int HKV_ = 4;
constexpr int HD_  = 128;
constexpr int FF_  = 8192;
constexpr int M_   = B_ * S_;          // 4096 rows
constexpr int KVW_ = HKV_ * HD_;       // 512
constexpr float SCALE_ = 0.08838834764831845f;  // 1/sqrt(128)

#if __has_builtin(__builtin_amdgcn_tensor_load_to_lds)
#define USE_TDM 1
#else
#define USE_TDM 0                      // host pass / toolchains without TDM builtin
#endif

// ---------------------------------------------------------------------------
// WMMA helpers (CDNA5: V_WMMA_F32_16X16X32_BF16, wave32)
// A-frag: lane L holds row M=L&15; K = (L>>4)*8 + {0..7} and 16+(L>>4)*8+{0..7}
// B fed as B^T rows (same loader).  C: v8f elem r -> row r / r+8, col = L&15.
// ---------------------------------------------------------------------------
DEV v16bf frag_ld(const bf16_t* __restrict__ base, int ld) {
  const int l = threadIdx.x & 31;
  const bf16_t* p = base + (size_t)(l & 15) * ld + (l >> 4) * 8;
  v8bf lo = *(const v8bf*)(p);
  v8bf hi = *(const v8bf*)(p + 16);
  return __builtin_shufflevector(lo, hi, 0,1,2,3,4,5,6,7,8,9,10,11,12,13,14,15);
}

DEV v8f wmma_bf16(v16bf a, v16bf b, v8f c) {
  return __builtin_amdgcn_wmma_f32_16x16x32_bf16(false, a, false, b, (short)0, c,
                                                 false, false);
}

DEV v8f v8f_zero() { v8f z = {0.f,0.f,0.f,0.f,0.f,0.f,0.f,0.f}; return z; }

// ---------------------------------------------------------------------------
// Tensor Data Mover: DMA a [rows=128, k=32] bf16 tile (row stride = ld elems)
// into LDS with 4-DWORD pad every 16 DWORDs -> LDS row stride 40 elems (80 B),
// matching frag_ld's padded layout.  Issued by one wave; TENSORcnt tracked.
// D# fields per CDNA5 ISA.
// ---------------------------------------------------------------------------
#if USE_TDM
DEV void tdm_load_tile(const bf16_t* gptr, unsigned lds_off, int dim0, int dim1,
                       long long stride_elems) {
  const unsigned long long ga = (unsigned long long)(size_t)gptr;
  u32x4 g0;
  g0[0] = 1u;                                            // count=1 (valid), user
  g0[1] = lds_off;                                       // lds_addr (bytes)
  g0[2] = (unsigned)(ga & 0xFFFFFFFFu);                  // global_addr[31:0]
  g0[3] = (unsigned)((ga >> 32) & 0x01FFFFFFu) | (2u << 30);  // addr hi | type=2
  i32x8 g1;
  g1[0] = (1 << 16) | (1 << 20) | (3 << 22) | (3 << 25); // 2B elems, pad: 4DW/16DW
  g1[1] = (dim0 & 0xFFFF) << 16;                         // tensor_dim0[15:0]
  g1[2] = ((dim0 >> 16) & 0xFFFF) | ((dim1 & 0xFFFF) << 16);
  g1[3] = ((dim1 >> 16) & 0xFFFF) | (32 << 16);          // tile_dim0 = 32
  g1[4] = 128;                                           // tile_dim1=128, dim2=0
  g1[5] = (int)(stride_elems & 0xFFFFFFFF);              // tensor_dim0_stride lo
  g1[6] = (int)((stride_elems >> 32) & 0xFFFF);          // stride hi | dim1_stride
  g1[7] = 0;
  i32x4 z4 = {0, 0, 0, 0};
#if defined(__clang_major__) && (__clang_major__ >= 23)
  i32x8 z8 = {0, 0, 0, 0, 0, 0, 0, 0};
  __builtin_amdgcn_tensor_load_to_lds(g0, g1, z4, z4, z8, 0);
#else
  __builtin_amdgcn_tensor_load_to_lds(g0, g1, z4, z4, 0);
#endif
}
#endif

DEV unsigned lds_offset_of(const void* p) { return (unsigned)(size_t)p; }

// Fallback staging: 256 threads copy a 128x32 bf16 tile into padded LDS.
DEV void stage_copy(bf16_t (*dst)[40], const bf16_t* __restrict__ src, int ld) {
  const int t = threadIdx.x;
  const int r = t >> 1, kh = (t & 1) * 16;
  const bf16_t* p = src + (size_t)r * ld + kh;
  *(v8bf*)&dst[r][kh]     = *(const v8bf*)(p);
  *(v8bf*)&dst[r][kh + 8] = *(const v8bf*)(p + 8);
}

// ---------------------------------------------------------------------------
// RMSNorm: one block per row, f32 in -> bf16 out
// ---------------------------------------------------------------------------
__global__ __launch_bounds__(256)
void rmsnorm_kernel(const float* __restrict__ x, const float* __restrict__ w,
                    bf16_t* __restrict__ out) {
  __shared__ float red[256];
  const int row = blockIdx.x;
  const float* xr = x + (size_t)row * D_;
  float ss = 0.f;
  for (int c = threadIdx.x; c < D_; c += 256) { float v = xr[c]; ss += v * v; }
  red[threadIdx.x] = ss;
  __syncthreads();
  for (int s = 128; s > 0; s >>= 1) {
    if (threadIdx.x < (unsigned)s) red[threadIdx.x] += red[threadIdx.x + s];
    __syncthreads();
  }
  const float inv = rsqrtf(red[0] * (1.0f / D_) + 1e-6f);
  bf16_t* orow = out + (size_t)row * D_;
  for (int c = threadIdx.x; c < D_; c += 256)
    orow[c] = (bf16_t)(xr[c] * inv * w[c]);
}

// ---------------------------------------------------------------------------
// Weight convert + transpose: W[K][N] f32 -> Wt[N][K] bf16 (32x32 LDS tiles)
// ---------------------------------------------------------------------------
__global__ __launch_bounds__(256)
void convT_kernel(const float* __restrict__ W, bf16_t* __restrict__ Wt,
                  int K, int N) {
  __shared__ float tile[32][33];
  const int n0 = blockIdx.x * 32, k0 = blockIdx.y * 32;
  const int tx = threadIdx.x & 31, ty = threadIdx.x >> 5;   // ty: 0..7
#pragma unroll
  for (int i = 0; i < 32; i += 8)
    tile[ty + i][tx] = W[(size_t)(k0 + ty + i) * N + n0 + tx];
  __syncthreads();
#pragma unroll
  for (int i = 0; i < 32; i += 8)
    Wt[(size_t)(n0 + ty + i) * K + k0 + tx] = (bf16_t)tile[tx][ty + i];
}

// ---------------------------------------------------------------------------
// WMMA GEMM: C[M,N] = A[M,K](bf16) * Bt[N,K](bf16, pre-transposed).
// Block tile 128x128, BK=32, double-buffered LDS fed by the TDM
// (tensor_load_to_lds) issued by wave 0; one barrier per K-step.
// 8 waves: 4(M) x 2(N); wave tile 32x64 -> 8 WMMAs / K-step / wave.
// ---------------------------------------------------------------------------
enum { EPI_BF16 = 0, EPI_RESID_F32 = 1, EPI_SILU_BF16 = 2,
       EPI_MUL_BF16 = 3, EPI_RESID_BIAS_F32 = 4 };

template <int EPI>
__global__ __launch_bounds__(256)
void gemm_kernel(const bf16_t* __restrict__ A, const bf16_t* __restrict__ Bt,
                 const float* __restrict__ bias, const float* __restrict__ resid,
                 const bf16_t* __restrict__ aux,
                 float* __restrict__ Cf, bf16_t* __restrict__ Cb,
                 int M, int N, int K) {
  __shared__ bf16_t As[2][128][40];   // [buf][m][k], +8 pad (TDM pad feature)
  __shared__ bf16_t Bs[2][128][40];   // [buf][n][k]

  const int t  = threadIdx.x;
  const int w  = t >> 5;
  const int wm = w & 3;
  const int wn = w >> 2;
  const int bm = blockIdx.y * 128;
  const int bn = blockIdx.x * 128;

  v8f acc[2][4];
#pragma unroll
  for (int mi = 0; mi < 2; ++mi)
#pragma unroll
    for (int ni = 0; ni < 4; ++ni) acc[mi][ni] = v8f_zero();

  const bf16_t* Abase = A  + (size_t)bm * K;
  const bf16_t* Bbase = Bt + (size_t)bn * K;
  const int iters = K / 32;
  int cur = 0;

#if USE_TDM
  if (w == 0) {
    tdm_load_tile(Abase, lds_offset_of(&As[0][0][0]), K, M - bm, K);
    tdm_load_tile(Bbase, lds_offset_of(&Bs[0][0][0]), K, N - bn, K);
  }
#else
  stage_copy(As[0], Abase, K);
  stage_copy(Bs[0], Bbase, K);
#endif

  for (int it = 0; it < iters; ++it) {
#if USE_TDM
    if (w == 0) __builtin_amdgcn_s_wait_tensorcnt(0);
#endif
    __syncthreads();   // tile `cur` ready for all waves; buf `cur^1` free
    if (it + 1 < iters) {
      const bf16_t* an = Abase + (size_t)(it + 1) * 32;
      const bf16_t* bn2 = Bbase + (size_t)(it + 1) * 32;
#if USE_TDM
      if (w == 0) {
        tdm_load_tile(an,  lds_offset_of(&As[cur ^ 1][0][0]), K, M - bm, K);
        tdm_load_tile(bn2, lds_offset_of(&Bs[cur ^ 1][0][0]), K, N - bn, K);
      }
#else
      stage_copy(As[cur ^ 1], an, K);
      stage_copy(Bs[cur ^ 1], bn2, K);
#endif
    }
    v16bf af[2], bfc[4];
#pragma unroll
    for (int mi = 0; mi < 2; ++mi)
      af[mi] = frag_ld(&As[cur][wm * 32 + mi * 16][0], 40);
#pragma unroll
    for (int ni = 0; ni < 4; ++ni)
      bfc[ni] = frag_ld(&Bs[cur][wn * 64 + ni * 16][0], 40);
#pragma unroll
    for (int mi = 0; mi < 2; ++mi)
#pragma unroll
      for (int ni = 0; ni < 4; ++ni)
        acc[mi][ni] = wmma_bf16(af[mi], bfc[ni], acc[mi][ni]);
    cur ^= 1;
  }

  // ---- epilogue ----
  const int lane = threadIdx.x & 31;
  const int lh = lane >> 4, ln = lane & 15;
#pragma unroll
  for (int mi = 0; mi < 2; ++mi)
#pragma unroll
    for (int ni = 0; ni < 4; ++ni)
#pragma unroll
      for (int rr = 0; rr < 8; ++rr) {
        const int row = bm + wm * 32 + mi * 16 + rr + 8 * lh;
        const int col = bn + wn * 64 + ni * 16 + ln;
        const size_t idx = (size_t)row * N + col;
        float v = acc[mi][ni][rr];
        if (EPI == EPI_BF16) {
          Cb[idx] = (bf16_t)v;
        } else if (EPI == EPI_RESID_F32) {
          Cf[idx] = resid[idx] + v;
        } else if (EPI == EPI_SILU_BF16) {
          v += bias[col];
          Cb[idx] = (bf16_t)(v / (1.0f + __expf(-v)));
        } else if (EPI == EPI_MUL_BF16) {
          v += bias[col];
          Cb[idx] = (bf16_t)(v * (float)aux[idx]);
        } else {  // EPI_RESID_BIAS_F32
          Cf[idx] = resid[idx] + v + bias[col];
        }
      }
}

// ---------------------------------------------------------------------------
// V transpose: v[B*S][KVW] -> vt[B][HKV][HD][S]  (for contiguous PV B-frags)
// ---------------------------------------------------------------------------
__global__ __launch_bounds__(256)
void vtrans_kernel(const bf16_t* __restrict__ v, bf16_t* __restrict__ vt) {
  const int idx = blockIdx.x * 256 + threadIdx.x;
  const int col  = idx & (KVW_ - 1);
  const int rowm = idx >> 9;
  const int b = rowm >> 11, s = rowm & (S_ - 1);
  const int g = col >> 7,  hd = col & (HD_ - 1);
  vt[((size_t)(b * HKV_ + g) * HD_ + hd) * S_ + s] = v[(size_t)rowm * KVW_ + col];
}

// ---------------------------------------------------------------------------
// Flash attention: block = (b, head h, 64 q-rows); 4 waves x 16 q-rows.
// S = (Q K^T)*scale + pos_bias + causal.  Online softmax, PV via WMMA.
// ---------------------------------------------------------------------------
__global__ __launch_bounds__(128)
void attn_kernel(const bf16_t* __restrict__ q, const bf16_t* __restrict__ k,
                 const bf16_t* __restrict__ vt, const float* __restrict__ pos_bias,
                 bf16_t* __restrict__ o) {
  __shared__ bf16_t Ps[4][16][40];   // per-wave P re-layout buffer

  const int b = blockIdx.z, h = blockIdx.y;
  const int g = h >> 2;                         // kv group (NREP=4)
  const int w = threadIdx.x >> 5;
  const int lane = threadIdx.x & 31;
  const int lh = lane >> 4, ln = lane & 15;
  const int q0 = blockIdx.x * 64 + w * 16;

  v16bf qf[4];
  const bf16_t* qbase = q + (size_t)(b * S_ + q0) * D_ + h * HD_;
#pragma unroll
  for (int kk = 0; kk < 4; ++kk) qf[kk] = frag_ld(qbase + kk * 32, D_);

  v8f accv[8];
  float m[8], l[8];
#pragma unroll
  for (int tt = 0; tt < 8; ++tt) accv[tt] = v8f_zero();
#pragma unroll
  for (int rr = 0; rr < 8; ++rr) { m[rr] = -3.0e38f; l[rr] = 0.f; }

  const int kmax = q0 + 15;                     // causal bound for this wave
  for (int k0 = 0; k0 <= kmax; k0 += 32) {
    if (k0 + 32 <= kmax)
      __builtin_prefetch(pos_bias + ((size_t)h * S_ + q0) * S_ + k0 + 32, 0, 0);
    float p[2][8];
    float rowmax[8];
#pragma unroll
    for (int rr = 0; rr < 8; ++rr) rowmax[rr] = -3.0e38f;

#pragma unroll
    for (int nt = 0; nt < 2; ++nt) {
      v8f s = v8f_zero();
      const bf16_t* kbase = k + (size_t)(b * S_ + k0 + nt * 16) * KVW_ + g * HD_;
#pragma unroll
      for (int kk = 0; kk < 4; ++kk)
        s = wmma_bf16(qf[kk], frag_ld(kbase + kk * 32, KVW_), s);
#pragma unroll
      for (int rr = 0; rr < 8; ++rr) {
        const int qrow = q0 + rr + 8 * lh;
        const int kcol = k0 + nt * 16 + ln;
        float sv = s[rr] * SCALE_ + pos_bias[((size_t)h * S_ + qrow) * S_ + kcol];
        sv = (kcol <= qrow) ? sv : -1e9f;
        p[nt][rr] = sv;
        rowmax[rr] = fmaxf(rowmax[rr], sv);
      }
    }
#pragma unroll
    for (int rr = 0; rr < 8; ++rr)
      for (int off = 8; off > 0; off >>= 1)
        rowmax[rr] = fmaxf(rowmax[rr], __shfl_xor(rowmax[rr], off, 16));

#pragma unroll
    for (int rr = 0; rr < 8; ++rr) {
      const float mnew = fmaxf(m[rr], rowmax[rr]);
      const float sc = __expf(m[rr] - mnew);
      m[rr] = mnew;
      float rsum = 0.f;
#pragma unroll
      for (int nt = 0; nt < 2; ++nt) {
        p[nt][rr] = __expf(p[nt][rr] - mnew);
        rsum += p[nt][rr];
      }
      for (int off = 8; off > 0; off >>= 1) rsum += __shfl_xor(rsum, off, 16);
      l[rr] = l[rr] * sc + rsum;
#pragma unroll
      for (int tt = 0; tt < 8; ++tt) accv[tt][rr] *= sc;
    }

    // re-layout P (C-layout) -> A-frag via wave-private LDS
#pragma unroll
    for (int rr = 0; rr < 8; ++rr)
#pragma unroll
      for (int nt = 0; nt < 2; ++nt)
        Ps[w][rr + 8 * lh][nt * 16 + ln] = (bf16_t)p[nt][rr];
    v16bf pf = frag_ld(&Ps[w][0][0], 40);

#pragma unroll
    for (int tt = 0; tt < 8; ++tt) {
      const bf16_t* vb = vt + ((size_t)(b * HKV_ + g) * HD_ + tt * 16) * S_ + k0;
      accv[tt] = wmma_bf16(pf, frag_ld(vb, S_), accv[tt]);
    }
  }

#pragma unroll
  for (int rr = 0; rr < 8; ++rr) l[rr] = 1.0f / l[rr];
#pragma unroll
  for (int tt = 0; tt < 8; ++tt)
#pragma unroll
    for (int rr = 0; rr < 8; ++rr) {
      const int row = b * S_ + q0 + rr + 8 * lh;
      const int col = h * HD_ + tt * 16 + ln;
      o[(size_t)row * D_ + col] = (bf16_t)(accv[tt][rr] * l[rr]);
    }
}

// ---------------------------------------------------------------------------
// Launcher
// ---------------------------------------------------------------------------
extern "C" void kernel_launch(void* const* d_in, const int* in_sizes, int n_in,
                              void* d_out, int out_size, void* d_ws, size_t ws_size,
                              hipStream_t stream) {
  (void)in_sizes; (void)n_in; (void)out_size; (void)ws_size;
  const float* x        = (const float*)d_in[0];
  const float* mask     = (const float*)d_in[1]; (void)mask;  // causal analytic
  const float* pos_bias = (const float*)d_in[2];
  const float* wq       = (const float*)d_in[3];
  const float* wk       = (const float*)d_in[4];
  const float* wv       = (const float*)d_in[5];
  const float* wo       = (const float*)d_in[6];
  const float* attn_nw  = (const float*)d_in[7];
  const float* lin_nw   = (const float*)d_in[8];
  const float* gate_w   = (const float*)d_in[9];
  const float* gate_b   = (const float*)d_in[10];
  const float* lin_w    = (const float*)d_in[11];
  const float* lin_b    = (const float*)d_in[12];
  const float* out_w    = (const float*)d_in[13];
  const float* out_b    = (const float*)d_in[14];
  float* out = (float*)d_out;

  char* ws = (char*)d_ws;
  size_t off = 0;
  auto alloc = [&](size_t bytes) -> void* {
    void* p = ws + off;
    off += (bytes + 255) & ~(size_t)255;
    return p;
  };
  bf16_t* xn = (bf16_t*)alloc((size_t)M_ * D_   * 2);
  bf16_t* qb = (bf16_t*)alloc((size_t)M_ * D_   * 2);
  bf16_t* kb = (bf16_t*)alloc((size_t)M_ * KVW_ * 2);
  bf16_t* vb = (bf16_t*)alloc((size_t)M_ * KVW_ * 2);
  bf16_t* vt = (bf16_t*)alloc((size_t)M_ * KVW_ * 2);
  bf16_t* ob = (bf16_t*)alloc((size_t)M_ * D_   * 2);
  float*  a  = (float*) alloc((size_t)M_ * D_   * 4);
  bf16_t* an = (bf16_t*)alloc((size_t)M_ * D_   * 2);
  bf16_t* hs = (bf16_t*)alloc((size_t)M_ * FF_  * 2);
  bf16_t* hh = (bf16_t*)alloc((size_t)M_ * FF_  * 2);
  bf16_t* wT = (bf16_t*)alloc((size_t)FF_ * D_  * 2);   // reused per weight

  // 1) attention pre-norm
  rmsnorm_kernel<<<M_, 256, 0, stream>>>(x, attn_nw, xn);

  // 2) q/k/v projections (weights transposed+bf16, then TDM-fed WMMA GEMM)
  convT_kernel<<<dim3(D_ / 32, D_ / 32), 256, 0, stream>>>(wq, wT, D_, D_);
  gemm_kernel<EPI_BF16><<<dim3(D_ / 128, M_ / 128), 256, 0, stream>>>(
      xn, wT, nullptr, nullptr, nullptr, nullptr, qb, M_, D_, D_);
  convT_kernel<<<dim3(KVW_ / 32, D_ / 32), 256, 0, stream>>>(wk, wT, D_, KVW_);
  gemm_kernel<EPI_BF16><<<dim3(KVW_ / 128, M_ / 128), 256, 0, stream>>>(
      xn, wT, nullptr, nullptr, nullptr, nullptr, kb, M_, KVW_, D_);
  convT_kernel<<<dim3(KVW_ / 32, D_ / 32), 256, 0, stream>>>(wv, wT, D_, KVW_);
  gemm_kernel<EPI_BF16><<<dim3(KVW_ / 128, M_ / 128), 256, 0, stream>>>(
      xn, wT, nullptr, nullptr, nullptr, nullptr, vb, M_, KVW_, D_);

  // 3) V transpose + flash attention
  vtrans_kernel<<<(M_ * KVW_) / 256, 256, 0, stream>>>(vb, vt);
  attn_kernel<<<dim3(S_ / 64, HQ_, B_), 128, 0, stream>>>(qb, kb, vt, pos_bias, ob);

  // 4) output projection + residual
  convT_kernel<<<dim3(D_ / 32, D_ / 32), 256, 0, stream>>>(wo, wT, D_, D_);
  gemm_kernel<EPI_RESID_F32><<<dim3(D_ / 128, M_ / 128), 256, 0, stream>>>(
      ob, wT, nullptr, x, nullptr, a, nullptr, M_, D_, D_);

  // 5) FFN pre-norm + SwiGLU + down projection
  rmsnorm_kernel<<<M_, 256, 0, stream>>>(a, lin_nw, an);
  convT_kernel<<<dim3(FF_ / 32, D_ / 32), 256, 0, stream>>>(lin_w, wT, D_, FF_);
  gemm_kernel<EPI_SILU_BF16><<<dim3(FF_ / 128, M_ / 128), 256, 0, stream>>>(
      an, wT, lin_b, nullptr, nullptr, nullptr, hs, M_, FF_, D_);
  convT_kernel<<<dim3(FF_ / 32, D_ / 32), 256, 0, stream>>>(gate_w, wT, D_, FF_);
  gemm_kernel<EPI_MUL_BF16><<<dim3(FF_ / 128, M_ / 128), 256, 0, stream>>>(
      an, wT, gate_b, nullptr, hs, nullptr, hh, M_, FF_, D_);
  convT_kernel<<<dim3(D_ / 32, FF_ / 32), 256, 0, stream>>>(out_w, wT, FF_, D_);
  gemm_kernel<EPI_RESID_BIAS_F32><<<dim3(D_ / 128, M_ / 128), 256, 0, stream>>>(
      hh, wT, out_b, a, nullptr, out, nullptr, M_, D_, FF_);
}
